// Attention_52802327937257
// MI455X (gfx1250) — compile-verified, matrix operand order checked by reading, not run on hardware
//
#include <hip/hip_runtime.h>

typedef unsigned short u16;
typedef __attribute__((ext_vector_type(8)))  float          v8f;
typedef __attribute__((ext_vector_type(16))) __bf16         v16bf;
typedef __attribute__((ext_vector_type(8)))  unsigned short v8u16;
typedef __attribute__((ext_vector_type(16))) unsigned short v16u16;
typedef __attribute__((ext_vector_type(4)))  int            v4i;

#define BDIM 8
#define SEQ  1024
#define CH   768
#define NH   12
#define HD   64
#define BH   (BDIM*NH)      // 96
#define MROWS (BDIM*SEQ)    // 8192
#define LDSS 48             // LDS row stride in bf16 elems (96B; fragment chunks stay 16B aligned)
#define KSTEPS (CH/32)      // 24
#define QSCALE (0.125f * 1.44269504088896f)  // head-dim scale * log2(e), pairs with exp2

// ---------------- CDNA5 async-to-LDS path (guarded; falls back to sync copies) ----------
#if defined(__HIP_DEVICE_COMPILE__) && __has_builtin(__builtin_amdgcn_global_load_async_to_lds_b128)
#define HAVE_ASYNC 1
#else
#define HAVE_ASYNC 0
#endif

static __device__ __forceinline__ void async_wait0() {
#if HAVE_ASYNC
#if __has_builtin(__builtin_amdgcn_s_wait_asynccnt)
    __builtin_amdgcn_s_wait_asynccnt(0);
#else
    asm volatile("s_wait_asynccnt 0x0" ::: "memory");
#endif
#endif
}

// copy one 32-byte (16 x bf16) row chunk global -> LDS
static __device__ __forceinline__ void stage_row32B(const u16* g, u16* l) {
#if HAVE_ASYNC
    __builtin_amdgcn_global_load_async_to_lds_b128(
        (__attribute__((address_space(1))) v4i*)(g),
        (__attribute__((address_space(3))) v4i*)(l), 0, 0);
    __builtin_amdgcn_global_load_async_to_lds_b128(
        (__attribute__((address_space(1))) v4i*)(g + 8),
        (__attribute__((address_space(3))) v4i*)(l + 8), 0, 0);
#else
    *(v8u16*)l       = *(const v8u16*)g;
    *(v8u16*)(l + 8) = *(const v8u16*)(g + 8);
#endif
}

// ---------------- helpers ----------------
static __device__ __forceinline__ u16 f2bf(float f) {
    unsigned u = __builtin_bit_cast(unsigned, f);
    unsigned r = u + 0x7FFFu + ((u >> 16) & 1u);   // round-to-nearest-even
    return (u16)(r >> 16);
}

// WMMA A/B 16-bit fragment: lane's two contiguous 8-elem chunks at offsets c0, c0+16
static __device__ __forceinline__ v16bf frag_ld(const u16* rowp, int c0) {
    v8u16 lo = *(const v8u16*)(rowp + c0);
    v8u16 hi = *(const v8u16*)(rowp + c0 + 16);
    v16u16 c = __builtin_shufflevector(lo, hi, 0,1,2,3,4,5,6,7,8,9,10,11,12,13,14,15);
    return __builtin_bit_cast(v16bf, c);
}

static __device__ __forceinline__ v8f wmma_bf16(v16bf a, v16bf b, v8f c) {
    return __builtin_amdgcn_wmma_f32_16x16x32_bf16(false, a, false, b, (short)0, c, false, false);
}

static __device__ __forceinline__ v8f vzero8() {
    return (v8f){0.f,0.f,0.f,0.f,0.f,0.f,0.f,0.f};
}

// ---------------- weight transpose + fp32->bf16 : dst[c*R + r] = bf16(src[r*C + c]) ----
__global__ __launch_bounds__(256) void wtrans_kernel(const float* __restrict__ src,
                                                     u16* __restrict__ dst, int R, int C) {
    __shared__ float t[32][33];
    int tx = threadIdx.x, ty = threadIdx.y;
    int x = blockIdx.x * 32 + tx;
    int y0 = blockIdx.y * 32;
#pragma unroll
    for (int i = 0; i < 4; ++i)
        t[ty + i * 8][tx] = src[(size_t)(y0 + ty + i * 8) * C + x];
    __syncthreads();
    int r = blockIdx.y * 32 + tx;
#pragma unroll
    for (int i = 0; i < 4; ++i)
        dst[(size_t)(blockIdx.x * 32 + ty + i * 8) * R + r] = f2bf(t[tx][ty + i * 8]);
}

// ---------------- QKV GEMM (double-buffered, async B staging) --------------------------
// X[8192,768]fp32 @ WqkvT[2304,768]bf16 -> Q(scaled)/K bf16 [bh][n][hd], Vt bf16 [bh][hd][n]
__global__ __launch_bounds__(256) void qkv_gemm_kernel(const float* __restrict__ X,
                                                       const u16* __restrict__ Wt,
                                                       u16* __restrict__ Qo,
                                                       u16* __restrict__ Ko,
                                                       u16* __restrict__ Vto) {
    __shared__ u16 As[2][128 * LDSS];
    __shared__ u16 Bs[2][128 * LDSS];
    const int m0 = blockIdx.y * 128;
    const int n0 = blockIdx.x * 128;
    const int tid = threadIdx.x;
    const int lane = tid & 31, wave = tid >> 5;
    const int wm = wave >> 1, wn = wave & 1;
    const int lr = lane & 15;
    const int c0 = (lane >> 4) * 8;
    const int hh = (lane >> 4) * 8;
    const int srow = tid >> 1, shalf = tid & 1;
    const int slot = srow * LDSS + shalf * 16;

    const float* abase = X  + (size_t)(m0 + srow) * CH + shalf * 16;
    const u16*   bbase = Wt + (size_t)(n0 + srow) * CH + shalf * 16;

    v8f acc[2][4];
#pragma unroll
    for (int i = 0; i < 2; ++i)
#pragma unroll
        for (int j = 0; j < 4; ++j) acc[i][j] = vzero8();

    // stage tile 0
    {
        float4 f0 = *(const float4*)(abase + 0);
        float4 f1 = *(const float4*)(abase + 4);
        float4 f2 = *(const float4*)(abase + 8);
        float4 f3 = *(const float4*)(abase + 12);
        v8u16 p0 = { f2bf(f0.x), f2bf(f0.y), f2bf(f0.z), f2bf(f0.w),
                     f2bf(f1.x), f2bf(f1.y), f2bf(f1.z), f2bf(f1.w) };
        v8u16 p1 = { f2bf(f2.x), f2bf(f2.y), f2bf(f2.z), f2bf(f2.w),
                     f2bf(f3.x), f2bf(f3.y), f2bf(f3.z), f2bf(f3.w) };
        *(v8u16*)&As[0][slot + 0] = p0;
        *(v8u16*)&As[0][slot + 8] = p1;
        stage_row32B(bbase, &Bs[0][slot]);
    }
    async_wait0();
    __syncthreads();

    for (int i = 0; i < KSTEPS; ++i) {
        const int cur = i & 1, nxt = cur ^ 1;
        const bool pf = (i + 1) < KSTEPS;
        float4 f0, f1, f2, f3;
        if (pf) {
            stage_row32B(bbase + (i + 1) * 32, &Bs[nxt][slot]);   // async: overlaps WMMA
            const float* ap = abase + (i + 1) * 32;
            f0 = *(const float4*)(ap + 0);
            f1 = *(const float4*)(ap + 4);
            f2 = *(const float4*)(ap + 8);
            f3 = *(const float4*)(ap + 12);
        }
        v16bf a[2], b[4];
#pragma unroll
        for (int tm = 0; tm < 2; ++tm)
            a[tm] = frag_ld(&As[cur][(wm * 32 + tm * 16 + lr) * LDSS], c0);
#pragma unroll
        for (int tn = 0; tn < 4; ++tn)
            b[tn] = frag_ld(&Bs[cur][(wn * 64 + tn * 16 + lr) * LDSS], c0);
#pragma unroll
        for (int tm = 0; tm < 2; ++tm)
#pragma unroll
            for (int tn = 0; tn < 4; ++tn)
                acc[tm][tn] = wmma_bf16(a[tm], b[tn], acc[tm][tn]);
        if (pf) {
            v8u16 p0 = { f2bf(f0.x), f2bf(f0.y), f2bf(f0.z), f2bf(f0.w),
                         f2bf(f1.x), f2bf(f1.y), f2bf(f1.z), f2bf(f1.w) };
            v8u16 p1 = { f2bf(f2.x), f2bf(f2.y), f2bf(f2.z), f2bf(f2.w),
                         f2bf(f3.x), f2bf(f3.y), f2bf(f3.z), f2bf(f3.w) };
            *(v8u16*)&As[nxt][slot + 0] = p0;
            *(v8u16*)&As[nxt][slot + 8] = p1;
        }
        async_wait0();
        __syncthreads();
    }

    // epilogue: `which` is uniform per block (768 = 6 * 128)
    const int which = n0 / CH;                 // 0=Q, 1=K, 2=V (scalar)
    const int ccb = n0 - which * CH + wn * 64;
#pragma unroll
    for (int tn = 0; tn < 4; ++tn) {
        const int cc = ccb + tn * 16 + lr;
        const int h = cc >> 6, hd = cc & 63;
#pragma unroll
        for (int tm = 0; tm < 2; ++tm)
#pragma unroll
            for (int j = 0; j < 8; ++j) {
                const int m = m0 + wm * 32 + tm * 16 + j + hh;
                const int bi = m >> 10, n = m & 1023;
                const int bh = bi * NH + h;
                const float v = acc[tm][tn][j];
                if (which == 0)      Qo[((size_t)bh * SEQ + n) * HD + hd] = f2bf(v * QSCALE);
                else if (which == 1) Ko[((size_t)bh * SEQ + n) * HD + hd] = f2bf(v);
                else                 Vto[((size_t)bh * HD + hd) * SEQ + n] = f2bf(v);
            }
    }
}

// ---------------- attention: per (bh, 128-row q block); each wave owns 16 q rows -------
__global__ __launch_bounds__(256) void attn_kernel(const u16* __restrict__ Q,
                                                   const u16* __restrict__ K,
                                                   const u16* __restrict__ Vt,
                                                   u16* __restrict__ AO) {
    __shared__ u16 Ps[8 * 16 * LDSS];
    const int bh = blockIdx.x;
    const int tid = threadIdx.x;
    const int lane = tid & 31, wave = tid >> 5;
    const int lr = lane & 15;
    const int c0 = (lane >> 4) * 8;
    const int hh = (lane >> 4) * 8;
    const int q0 = blockIdx.y * 128 + wave * 16;
    u16* pbuf = &Ps[wave * 16 * LDSS];

    v16bf qf[2];
#pragma unroll
    for (int ks = 0; ks < 2; ++ks)
        qf[ks] = frag_ld(Q + ((size_t)bh * SEQ + q0 + lr) * HD + ks * 32, c0);

    float mrow[8], lrow[8];
#pragma unroll
    for (int j = 0; j < 8; ++j) { mrow[j] = -1.0e30f; lrow[j] = 0.f; }
    v8f o[4];
#pragma unroll
    for (int tn = 0; tn < 4; ++tn) o[tn] = vzero8();

    for (int kc = 0; kc < SEQ; kc += 32) {
        v8f s[2];
        s[0] = vzero8(); s[1] = vzero8();
#pragma unroll
        for (int t = 0; t < 2; ++t)
#pragma unroll
            for (int ks = 0; ks < 2; ++ks) {
                v16bf kf = frag_ld(K + ((size_t)bh * SEQ + kc + t * 16 + lr) * HD + ks * 32, c0);
                s[t] = wmma_bf16(qf[ks], kf, s[t]);
            }
        // online softmax in base-2 domain (log2e folded into Q scale)
#pragma unroll
        for (int j = 0; j < 8; ++j) {
            float mx = fmaxf(s[0][j], s[1][j]);
#pragma unroll
            for (int off = 1; off <= 8; off <<= 1)
                mx = fmaxf(mx, __shfl_xor(mx, off, 32));
            float mn = fmaxf(mrow[j], mx);
            float corr = __builtin_amdgcn_exp2f(mrow[j] - mn);
            mrow[j] = mn;
            float p0 = __builtin_amdgcn_exp2f(s[0][j] - mn);
            float p1 = __builtin_amdgcn_exp2f(s[1][j] - mn);
            float rs = p0 + p1;
#pragma unroll
            for (int off = 1; off <= 8; off <<= 1)
                rs += __shfl_xor(rs, off, 32);
            lrow[j] = lrow[j] * corr + rs;
#pragma unroll
            for (int tn = 0; tn < 4; ++tn) o[tn][j] *= corr;
            pbuf[(j + hh) * LDSS + lr]      = f2bf(p0);
            pbuf[(j + hh) * LDSS + lr + 16] = f2bf(p1);
        }
        v16bf pf = frag_ld(pbuf + lr * LDSS, c0);
#pragma unroll
        for (int tn = 0; tn < 4; ++tn) {
            v16bf vf = frag_ld(Vt + ((size_t)bh * HD + tn * 16 + lr) * SEQ + kc, c0);
            o[tn] = wmma_bf16(pf, vf, o[tn]);
        }
    }

    const int b = bh / NH, h = bh % NH;
#pragma unroll
    for (int tn = 0; tn < 4; ++tn)
#pragma unroll
        for (int j = 0; j < 8; ++j) {
            const int n = q0 + j + hh;
            AO[((size_t)b * SEQ + n) * CH + h * HD + tn * 16 + lr] = f2bf(o[tn][j] / lrow[j]);
        }
}

// ---------------- proj GEMM (double-buffered, fully async staging) ---------------------
// AO[8192,768]bf16 @ WprojT[768,768]bf16 + bias -> fp32
__global__ __launch_bounds__(256) void proj_gemm_kernel(const u16* __restrict__ A,
                                                        const u16* __restrict__ Wt,
                                                        const float* __restrict__ bias,
                                                        float* __restrict__ Out) {
    __shared__ u16 As[2][128 * LDSS];
    __shared__ u16 Bs[2][128 * LDSS];
    const int m0 = blockIdx.y * 128;
    const int n0 = blockIdx.x * 128;
    const int tid = threadIdx.x;
    const int lane = tid & 31, wave = tid >> 5;
    const int wm = wave >> 1, wn = wave & 1;
    const int lr = lane & 15;
    const int c0 = (lane >> 4) * 8;
    const int hh = (lane >> 4) * 8;
    const int srow = tid >> 1, shalf = tid & 1;
    const int slot = srow * LDSS + shalf * 16;

    const u16* abase = A  + (size_t)(m0 + srow) * CH + shalf * 16;
    const u16* bbase = Wt + (size_t)(n0 + srow) * CH + shalf * 16;

    v8f acc[2][4];
#pragma unroll
    for (int i = 0; i < 2; ++i)
#pragma unroll
        for (int j = 0; j < 4; ++j) acc[i][j] = vzero8();

    stage_row32B(abase, &As[0][slot]);
    stage_row32B(bbase, &Bs[0][slot]);
    async_wait0();
    __syncthreads();

    for (int i = 0; i < KSTEPS; ++i) {
        const int cur = i & 1, nxt = cur ^ 1;
        if ((i + 1) < KSTEPS) {
            stage_row32B(abase + (i + 1) * 32, &As[nxt][slot]);
            stage_row32B(bbase + (i + 1) * 32, &Bs[nxt][slot]);
        }
        v16bf a[2], b[4];
#pragma unroll
        for (int tm = 0; tm < 2; ++tm)
            a[tm] = frag_ld(&As[cur][(wm * 32 + tm * 16 + lr) * LDSS], c0);
#pragma unroll
        for (int tn = 0; tn < 4; ++tn)
            b[tn] = frag_ld(&Bs[cur][(wn * 64 + tn * 16 + lr) * LDSS], c0);
#pragma unroll
        for (int tm = 0; tm < 2; ++tm)
#pragma unroll
            for (int tn = 0; tn < 4; ++tn)
                acc[tm][tn] = wmma_bf16(a[tm], b[tn], acc[tm][tn]);
        async_wait0();
        __syncthreads();
    }

#pragma unroll
    for (int tm = 0; tm < 2; ++tm)
#pragma unroll
        for (int tn = 0; tn < 4; ++tn)
#pragma unroll
            for (int j = 0; j < 8; ++j) {
                const int m = m0 + wm * 32 + tm * 16 + j + hh;
                const int ncol = n0 + wn * 64 + tn * 16 + lr;
                Out[(size_t)m * CH + ncol] = acc[tm][tn][j] + bias[ncol];
            }
}

extern "C" void kernel_launch(void* const* d_in, const int* in_sizes, int n_in,
                              void* d_out, int out_size, void* d_ws, size_t ws_size,
                              hipStream_t stream) {
    const float* x     = (const float*)d_in[0];   // [8,1024,768]
    const float* wqkv  = (const float*)d_in[1];   // [768, 2304]
    const float* wproj = (const float*)d_in[2];   // [768, 768]
    const float* bproj = (const float*)d_in[3];   // [768]
    float* out = (float*)d_out;

    char* ws = (char*)d_ws;
    size_t off = 0;
    u16* wqkvT  = (u16*)(ws + off); off += (size_t)3 * CH * CH * 2;          // [2304][768]
    u16* wprojT = (u16*)(ws + off); off += (size_t)CH * CH * 2;              // [768][768]
    u16* Qb     = (u16*)(ws + off); off += (size_t)BH * SEQ * HD * 2;        // [bh][n][hd]
    u16* Kb     = (u16*)(ws + off); off += (size_t)BH * SEQ * HD * 2;        // [bh][n][hd]
    u16* Vtb    = (u16*)(ws + off); off += (size_t)BH * HD * SEQ * 2;        // [bh][hd][n]
    u16* AOb    = (u16*)(ws + off); off += (size_t)MROWS * CH * 2;           // [8192][768]
    (void)ws_size; (void)in_sizes; (void)n_in; (void)out_size;

    wtrans_kernel<<<dim3((3 * CH) / 32, CH / 32), dim3(32, 8), 0, stream>>>(wqkv, wqkvT, CH, 3 * CH);
    wtrans_kernel<<<dim3(CH / 32, CH / 32),       dim3(32, 8), 0, stream>>>(wproj, wprojT, CH, CH);
    qkv_gemm_kernel<<<dim3((3 * CH) / 128, MROWS / 128), 256, 0, stream>>>(x, wqkvT, Qb, Kb, Vtb);
    attn_kernel<<<dim3(BH, SEQ / 128), 256, 0, stream>>>(Qb, Kb, Vtb, AOb);
    proj_gemm_kernel<<<dim3(CH / 128, MROWS / 128), 256, 0, stream>>>(AOb, wprojT, bproj, out);
}
